// MultiHeadAttention_85177791414467
// MI455X (gfx1250) — matrix operand order repaired and checked by reading, not compiled, not run on hardware
//
#include <hip/hip_runtime.h>

typedef _Float16 h16;
typedef __attribute__((ext_vector_type(16))) _Float16 v16h;
typedef __attribute__((ext_vector_type(8)))  float    v8f;

#define WMMA_F32_F16(a, b, c) \
  __builtin_amdgcn_wmma_f32_16x16x32_f16(false, (a), false, (b), (short)0, (c), false, false)

union Frag16 { v16h v; uint32_t u[8]; };
union H4     { uint2 u2; h16 h[4]; };

// ---------------------------------------------------------------------------
// Tiled WMMA GEMM: C[M x N] = A[M x 1024] * W[1024 x N] + bias
// Block tile 128x128, BK=64, 128 threads (4 waves, 2x2), wave tile 64x64.
// 16 WMMAs per 16 ds_load_b128 per sub-step (1.0 loads/WMMA).
// QKV=true : A is f32 (x); epilogue splits columns into Q(scaled)/K/V f16 ws.
// QKV=false: A is f16 (attention output); epilogue writes f32 d_out.
// ---------------------------------------------------------------------------
template <bool QKV>
__global__ __launch_bounds__(128) void gemm_wmma(
    const float* __restrict__ Af32, const h16* __restrict__ Af16,
    const float* __restrict__ W, const float* __restrict__ bias,
    h16* __restrict__ qd, h16* __restrict__ kd, h16* __restrict__ vd,
    float* __restrict__ outd, int N)
{
  __shared__ __align__(16) h16 As[128 * 64];  // [row][k]
  __shared__ __align__(16) h16 Bs[128 * 64];  // [n][k] (transposed)

  const int tid  = threadIdx.x;
  const int lane = tid & 31;
  const int wid  = tid >> 5;
  const int wm   = wid >> 1;  // 0..1  (M direction)
  const int wn   = wid & 1;   // 0..1  (N direction)
  const int m0   = blockIdx.x * 128;
  const int n0   = blockIdx.y * 128;

  v8f acc[4][4] = {};

  for (int k0 = 0; k0 < 1024; k0 += 64) {
    __syncthreads();
    // ---- stage A tile 128x64 (4 elements per thread per step) ----
#pragma unroll
    for (int r = 0; r < 16; ++r) {
      int e = (r * 128 + tid) * 4;     // flattened element base
      int ar = e >> 6, ac = e & 63;
      H4 hv;
      if (QKV) {
        float4 f = *(const float4*)(Af32 + (size_t)(m0 + ar) * 1024 + k0 + ac);
        hv.h[0] = (h16)f.x; hv.h[1] = (h16)f.y; hv.h[2] = (h16)f.z; hv.h[3] = (h16)f.w;
      } else {
        hv.u2 = *(const uint2*)(Af16 + (size_t)(m0 + ar) * 1024 + k0 + ac);
      }
      *(uint2*)(As + ar * 64 + ac) = hv.u2;
    }
    // ---- stage B tile 64x128 transposed: Bs[n][k] = W[k][n] ----
#pragma unroll
    for (int r = 0; r < 16; ++r) {
      int e = (r * 128 + tid) * 4;
      int kr = e >> 7, nc = e & 127;
      float4 f = *(const float4*)(W + (size_t)(k0 + kr) * N + n0 + nc);
      Bs[(nc + 0) * 64 + kr] = (h16)f.x;
      Bs[(nc + 1) * 64 + kr] = (h16)f.y;
      Bs[(nc + 2) * 64 + kr] = (h16)f.z;
      Bs[(nc + 3) * 64 + kr] = (h16)f.w;
    }
    // prefetch next tiles while this chunk computes
    if (k0 + 64 < 1024) {
      int e = tid * 4;
      if (QKV) __builtin_prefetch(Af32 + (size_t)(m0 + (e >> 6)) * 1024 + k0 + 64 + (e & 63), 0, 3);
      else     __builtin_prefetch(Af16 + (size_t)(m0 + (e >> 6)) * 1024 + k0 + 64 + (e & 63), 0, 3);
      __builtin_prefetch(W + (size_t)(k0 + 64 + (e >> 7)) * N + n0 + (e & 127), 0, 3);
    }
    __syncthreads();

    // ---- compute: two 32-deep WMMA sub-steps, 16 WMMAs each ----
#pragma unroll
    for (int kk = 0; kk < 64; kk += 32) {
      Frag16 af[4], bf[4];
#pragma unroll
      for (int mt = 0; mt < 4; ++mt) {
        int row = wm * 64 + mt * 16 + (lane & 15);
        int kb = (lane >> 4) * 8;
#pragma unroll
        for (int jp = 0; jp < 8; ++jp) {
          int j = jp * 2;
          int k = ((j >> 3) << 4) + kb + (j & 7);
          af[mt].u[jp] = *(const uint32_t*)(As + row * 64 + kk + k);
        }
      }
#pragma unroll
      for (int nt = 0; nt < 4; ++nt) {
        int col = wn * 64 + nt * 16 + (lane & 15);
        int kb = (lane >> 4) * 8;
#pragma unroll
        for (int jp = 0; jp < 8; ++jp)
          bf[nt].u[jp] = *(const uint32_t*)(Bs + col * 64 + kk + (((jp * 2) >> 3) << 4) + kb + ((jp * 2) & 7));
      }
#pragma unroll
      for (int mt = 0; mt < 4; ++mt)
#pragma unroll
        for (int nt = 0; nt < 4; ++nt)
          acc[mt][nt] = WMMA_F32_F16(af[mt].v, bf[nt].v, acc[mt][nt]);
    }
  }

  // ---- epilogue (C layout: vgpr v -> row v + 8*(lane/16), col lane%16) ----
#pragma unroll
  for (int nt = 0; nt < 4; ++nt) {
    int ntile = n0 + wn * 64 + nt * 16;           // wave-uniform tile base
    int n = ntile + (lane & 15);
    float bv = bias[n];
    if (QKV) {
      // 16-wide tile never crosses Q/K/V (1024) or head (64) boundaries
      int three = ntile >> 10;
      int h     = (ntile >> 6) & 15;
      int dh    = n & 63;
      h16* dst  = (three == 0) ? qd : (three == 1) ? kd : vd;
      float scl = (three == 0) ? 0.125f : 1.0f;   // DH^-0.5 folded into Q
#pragma unroll
      for (int mt = 0; mt < 4; ++mt)
#pragma unroll
        for (int vv = 0; vv < 8; ++vv) {
          int mrow = m0 + wm * 64 + mt * 16 + vv + ((lane >> 4) << 3);
          int b = mrow >> 11, ss = mrow & 2047;
          size_t di = (((size_t)(b * 16 + h)) * 2048 + ss) * 64 + dh;
          dst[di] = (h16)((acc[mt][nt][vv] + bv) * scl);
        }
    } else {
#pragma unroll
      for (int mt = 0; mt < 4; ++mt)
#pragma unroll
        for (int vv = 0; vv < 8; ++vv) {
          int mrow = m0 + wm * 64 + mt * 16 + vv + ((lane >> 4) << 3);
          outd[(size_t)mrow * 1024 + n] = acc[mt][nt][vv] + bv;
        }
    }
  }
}

// ---------------------------------------------------------------------------
// Flash attention: grid (B*H, S/128), 256 threads = 8 waves, 16 q-rows/wave.
// Q pre-scaled. Row-sum of P obtained via a 5th WMMA against an all-ones
// B operand (replaces the shuffle-add reduction); only row-max uses shuffles.
// ---------------------------------------------------------------------------
__global__ __launch_bounds__(256) void attn_wmma(
    const h16* __restrict__ qg, const h16* __restrict__ kg,
    const h16* __restrict__ vg, h16* __restrict__ og)
{
  __shared__ __align__(16) h16 kbuf[32 * 64];     // [key][d]
  __shared__ __align__(16) h16 vtbuf[64 * 32];    // [d][key] (transposed)
  __shared__ __align__(16) h16 pbuf[8][16 * 32];  // per-wave P re-layout

  const int tid  = threadIdx.x;
  const int lane = tid & 31;
  const int wid  = tid >> 5;
  const int bh   = blockIdx.x;                    // b*16 + h
  const int qrow0 = blockIdx.y * 128 + wid * 16;
  const size_t base = (size_t)bh * 2048 * 64;

  // Q fragments (A-matrix layout), held in registers for entire kernel
  Frag16 qf[2];
#pragma unroll
  for (int f = 0; f < 2; ++f) {
    int row = qrow0 + (lane & 15);
    int kb = (lane >> 4) * 8;
#pragma unroll
    for (int jp = 0; jp < 8; ++jp) {
      int j = jp * 2;
      int k = f * 32 + ((j >> 3) << 4) + kb + (j & 7);
      qf[f].u[jp] = *(const uint32_t*)(qg + base + (size_t)row * 64 + k);
    }
  }

  // all-ones B operand for the row-sum WMMA (f16 1.0 = 0x3C00)
  Frag16 ones;
#pragma unroll
  for (int jp = 0; jp < 8; ++jp) ones.u[jp] = 0x3C003C00u;

  float m[8];
  v8f o[4] = {};
  v8f osum = {};   // running softmax denominators (replicated across 16 cols)
#pragma unroll
  for (int i = 0; i < 8; ++i) m[i] = -1e30f;

  for (int kb32 = 0; kb32 < 2048; kb32 += 32) {
    __syncthreads();
    {  // stage K [32][64]: 256 uint4 over 256 threads -> 1 each
      const uint4* src = (const uint4*)(kg + base + (size_t)kb32 * 64);
      uint4* dst = (uint4*)kbuf;
      dst[tid] = src[tid];
      // stage V transposed [d][key]: one uint4 (8 halves, same key row) per thread
      const h16* vsrc = vg + base + (size_t)kb32 * 64;
      H4 a, b;
      {
        uint4 w = *(const uint4*)(vsrc + tid * 8);
        a.u2 = make_uint2(w.x, w.y);
        b.u2 = make_uint2(w.z, w.w);
      }
      int key = tid >> 3;             // 64 halves per key row, 8 per thread
      int d0  = (tid & 7) * 8;
#pragma unroll
      for (int i = 0; i < 4; ++i) {
        vtbuf[(d0 + i) * 32 + key]     = a.h[i];
        vtbuf[(d0 + 4 + i) * 32 + key] = b.h[i];
      }
    }
    if (kb32 + 32 < 2048) {
      __builtin_prefetch(kg + base + (size_t)(kb32 + 32) * 64 + tid * 8, 0, 3);
      __builtin_prefetch(vg + base + (size_t)(kb32 + 32) * 64 + tid * 8, 0, 3);
    }
    __syncthreads();

    // ---- scores for two 16-key tiles ----
    v8f c0 = {}, c1 = {};
#pragma unroll
    for (int f = 0; f < 2; ++f) {
      Frag16 kf0, kf1;
      int kbb = (lane >> 4) * 8;
#pragma unroll
      for (int jp = 0; jp < 8; ++jp) {
        int d = f * 32 + (((jp * 2) >> 3) << 4) + kbb + ((jp * 2) & 7);
        kf0.u[jp] = *(const uint32_t*)(kbuf + (lane & 15) * 64 + d);
        kf1.u[jp] = *(const uint32_t*)(kbuf + (16 + (lane & 15)) * 64 + d);
      }
      c0 = WMMA_F32_F16(qf[f].v, kf0.v, c0);
      c1 = WMMA_F32_F16(qf[f].v, kf1.v, c1);
    }

    // ---- online softmax: row-max butterfly (row lives in one 16-lane half) ----
    float p0[8], p1[8], sc[8];
#pragma unroll
    for (int vv = 0; vv < 8; ++vv) {
      float x = fmaxf(c0[vv], c1[vv]);
#pragma unroll
      for (int off = 8; off >= 1; off >>= 1) x = fmaxf(x, __shfl_xor(x, off));
      float nm = fmaxf(m[vv], x);
      sc[vv] = __expf(m[vv] - nm);
      m[vv] = nm;
      p0[vv] = __expf(c0[vv] - nm);
      p1[vv] = __expf(c1[vv] - nm);
    }
#pragma unroll
    for (int vv = 0; vv < 8; ++vv) {
      osum[vv] = osum[vv] * sc[vv];
#pragma unroll
      for (int dt = 0; dt < 4; ++dt) o[dt][vv] = o[dt][vv] * sc[vv];
    }

    // ---- P (C layout) -> LDS -> A-fragment layout ----
    h16* pb = pbuf[wid];
#pragma unroll
    for (int vv = 0; vv < 8; ++vv) {
      int rv = vv + ((lane >> 4) << 3);
      pb[rv * 32 + (lane & 15)]      = (h16)p0[vv];
      pb[rv * 32 + 16 + (lane & 15)] = (h16)p1[vv];
    }
    __syncthreads();

    Frag16 pa;
    {
      int row = lane & 15;
      int kb = (lane >> 4) * 8;
#pragma unroll
      for (int jp = 0; jp < 8; ++jp) {
        int j = jp * 2;
        int k = ((j >> 3) << 4) + kb + (j & 7);
        pa.u[jp] = *(const uint32_t*)(pb + row * 32 + k);
      }
    }
#pragma unroll
    for (int dt = 0; dt < 4; ++dt) {
      Frag16 vf;
      int kbb = (lane >> 4) * 8;
#pragma unroll
      for (int jp = 0; jp < 8; ++jp)
        vf.u[jp] = *(const uint32_t*)(vtbuf + (dt * 16 + (lane & 15)) * 32 + (((jp * 2) >> 3) << 4) + kbb + ((jp * 2) & 7));
      o[dt] = WMMA_F32_F16(pa.v, vf.v, o[dt]);
    }
    // row-sum accumulation: P x ones -> every column holds sum_k P[q,k]
    osum = WMMA_F32_F16(pa.v, ones.v, osum);
  }

  // ---- finalize: divide by denominators, write f16 [8192][1024] ----
  const int b = bh >> 4, h = bh & 15;
#pragma unroll
  for (int vv = 0; vv < 8; ++vv) {
    float inv = 1.0f / osum[vv];
    size_t orow = (size_t)b * 2048 + qrow0 + vv + ((lane >> 4) << 3);
#pragma unroll
    for (int dt = 0; dt < 4; ++dt)
      og[orow * 1024 + h * 64 + dt * 16 + (lane & 15)] = (h16)(o[dt][vv] * inv);
  }
}

// ---------------------------------------------------------------------------
extern "C" void kernel_launch(void* const* d_in, const int* in_sizes, int n_in,
                              void* d_out, int out_size, void* d_ws, size_t ws_size,
                              hipStream_t stream) {
  const float* x     = (const float*)d_in[0];
  const float* w_qkv = (const float*)d_in[1];
  const float* b_qkv = (const float*)d_in[2];
  const float* w_out = (const float*)d_in[3];
  const float* b_out = (const float*)d_in[4];
  float* out = (float*)d_out;

  const size_t NQ = (size_t)8192 * 1024;
  h16* qws = (h16*)d_ws;
  h16* kws = qws + NQ;
  h16* vws = kws + NQ;
  h16* aws = vws + NQ;  // total 64 MB f16 workspace

  // QKV projection: M=8192, N=3072
  gemm_wmma<true><<<dim3(64, 24), 128, 0, stream>>>(
      x, nullptr, w_qkv, b_qkv, qws, kws, vws, nullptr, 3072);
  // Flash attention: 64 heads x 16 query blocks (128 q-rows/block)
  attn_wmma<<<dim3(64, 16), 256, 0, stream>>>(qws, kws, vws, aws);
  // Output projection: M=8192, N=1024
  gemm_wmma<false><<<dim3(64, 8), 128, 0, stream>>>(
      nullptr, aws, w_out, b_out, nullptr, nullptr, nullptr, out, 1024);
}
